// LinearLayer_PSLoRA_63677185131155
// MI455X (gfx1250) — matrix operand order repaired and checked by reading, not compiled, not run on hardware
//
#include <hip/hip_runtime.h>
#include <cstdint>

// ---------------- CDNA5 (gfx1250) fused LoRA linear ----------------
// out = X @ W^T + bias + 0.5 * ((X @ A[labeler]) @ B_lora^T)
// M = B*S = 16384, K = 4096, N = 4096, R = 32
// bf16 WMMA (v_wmma_f32_16x16x32_bf16), f32 accumulate.
// Block tile 256x128, wave tile 64x64 (4x4 fragments), double-buffered LDS
// with register-staged global->LDS software pipeline.

typedef __attribute__((ext_vector_type(16))) __bf16        v16bf;
typedef __attribute__((ext_vector_type(2)))  __bf16        v2bf;
typedef __attribute__((ext_vector_type(8)))  float         v8f;
typedef __attribute__((ext_vector_type(4)))  unsigned int  u32x4;
typedef __attribute__((ext_vector_type(2)))  unsigned int  u32x2;
typedef __attribute__((ext_vector_type(4)))  float         f32x4;

#define DIN    4096
#define DOUT   4096
#define SEQ    2048
#define BATCH  8
#define RNK    32
#define LSCALE 0.5f

#define BLK_M 256
#define BLK_N 128
#define BLK_K 32
#define LDX   40   // LDS row stride in halves (32 data + 8 pad) -> 80B, 16B-aligned

#if defined(__has_builtin)
#if __has_builtin(__builtin_amdgcn_cvt_pk_bf16_f32)
#define HAVE_PK_BF16 1
#endif
#endif

// round-to-nearest-even f32 -> bf16 (bit pattern)
__device__ __forceinline__ uint32_t bf16_bits(float x) {
  uint32_t u = __float_as_uint(x);
  return (u + 0x7FFFu + ((u >> 16) & 1u)) >> 16;
}
__device__ __forceinline__ uint32_t bf16_pack2(float lo, float hi) {
#ifdef HAVE_PK_BF16
  v2bf r = __builtin_amdgcn_cvt_pk_bf16_f32(lo, hi);   // v_cvt_pk_bf16_f32
  return __builtin_bit_cast(uint32_t, r);
#else
  return bf16_bits(lo) | (bf16_bits(hi) << 16);
#endif
}

union Frag16 { v16bf v; u32x4 q[2]; };

// A-operand (16x32 bf16): lane&15 = row M; lanes 0-15 hold K{0..7,16..23},
// lanes 16-31 hold K{8..15,24..31}.  base points at row 0 of the 16-row group.
__device__ __forceinline__ v16bf load_a_frag(const uint16_t* base, int lane) {
  const uint16_t* p = base + (lane & 15) * LDX + ((lane & 16) ? 8 : 0);
  Frag16 f;
  f.q[0] = *(const u32x4*)(p);
  f.q[1] = *(const u32x4*)(p + 16);
  return f.v;
}

// B-operand (32x16 bf16): lane&15 = column n; lanes 0-15 hold K{0..7,16..23},
// lanes 16-31 hold K{8..15,24..31}.  LDS is n-major so K is contiguous per lane.
__device__ __forceinline__ v16bf load_b_frag(const uint16_t* base, int lane) {
  const uint16_t* p = base + (lane & 15) * LDX + ((lane & 16) ? 8 : 0);
  Frag16 f;
  f.q[0] = *(const u32x4*)(p);
  f.q[1] = *(const u32x4*)(p + 16);
  return f.v;
}

__global__ __launch_bounds__(256)
void lora_linear_wmma_kernel(const float* __restrict__ X,
                             const float* __restrict__ W,
                             const float* __restrict__ bias,
                             const float* __restrict__ loraA,
                             const float* __restrict__ loraB,
                             const int*   __restrict__ labeler,
                             float* __restrict__ out)
{
  __shared__ __align__(16) uint16_t Xs [2][BLK_M * LDX];  // X tile, m-major
  __shared__ __align__(16) uint16_t Ws [2][BLK_N * LDX];  // W tile, n-major
  __shared__ __align__(16) uint16_t AsT[2][RNK   * LDX];  // loraA tile transposed [r][k]
  __shared__ __align__(16) uint16_t Bls[BLK_N * LDX];     // loraB tile, n-major over r
  __shared__ __align__(16) uint16_t AXs[BLK_M * LDX];     // scaled ax, m-major over r

  const int tid  = threadIdx.x;
  const int lane = tid & 31;
  const int wave = tid >> 5;
  const int mG   = wave >> 1;     // 0..3 -> 64 rows each
  const int nG   = wave & 1;      // 0..1 -> 64 cols each

  const int m0 = blockIdx.y * BLK_M;
  const int n0 = blockIdx.x * BLK_N;

  // One labeler per block: SEQ (2048) is a multiple of BLK_M (256).
  const int lab = labeler[m0 / SEQ];
  const float* Alab = loraA + (size_t)lab * DIN * RNK;

  // Per-thread staging coordinates (hoisted out of the K loop).
  const int srow = tid >> 3;            // 0..31
  const int sc4  = (tid & 7) << 2;      // 0..28

  // ---- stage loraB tile once: rows n0..n0+127, 32 f32 each -> bf16 ----
  #pragma unroll
  for (int i = 0; i < 4; ++i) {
    int row = srow + i * 32;
    f32x4 f = *(const f32x4*)(loraB + (size_t)(n0 + row) * RNK + sc4);
    u32x2 p; p.x = bf16_pack2(f.x, f.y); p.y = bf16_pack2(f.z, f.w);
    *(u32x2*)&Bls[row * LDX + sc4] = p;
  }

  v8f acc[4][4];
  v8f axa[4][2];
  {
    v8f z = {0.f,0.f,0.f,0.f,0.f,0.f,0.f,0.f};
    #pragma unroll
    for (int i = 0; i < 4; ++i) {
      #pragma unroll
      for (int j = 0; j < 4; ++j) acc[i][j] = z;
      axa[i][0] = z; axa[i][1] = z;
    }
  }

  // Hoisted per-thread global pointers; advance by BLK_K each iteration.
  const float* xg = X    + (size_t)(m0 + srow) * DIN + sc4;  // +i*32 rows
  const float* wg = W    + (size_t)(n0 + srow) * DIN + sc4;  // +i*32 rows
  const float* ag = Alab + (size_t)srow * RNK + sc4;

  // ---- pipeline prologue: issue global loads for K-tile 0 into registers ----
  f32x4 rx[8], rw[4], ra;
  #pragma unroll
  for (int i = 0; i < 8; ++i) rx[i] = *(const f32x4*)(xg + (size_t)(i * 32) * DIN);
  #pragma unroll
  for (int i = 0; i < 4; ++i) rw[i] = *(const f32x4*)(wg + (size_t)(i * 32) * DIN);
  ra = *(const f32x4*)(ag);

  for (int k0 = 0; k0 < DIN; k0 += BLK_K) {
    const int buf = (k0 >> 5) & 1;

    // ---- convert + store the register-staged tile into LDS[buf] ----
    #pragma unroll
    for (int i = 0; i < 8; ++i) {
      u32x2 p; p.x = bf16_pack2(rx[i].x, rx[i].y); p.y = bf16_pack2(rx[i].z, rx[i].w);
      *(u32x2*)&Xs[buf][(srow + i * 32) * LDX + sc4] = p;
    }
    #pragma unroll
    for (int i = 0; i < 4; ++i) {
      u32x2 p; p.x = bf16_pack2(rw[i].x, rw[i].y); p.y = bf16_pack2(rw[i].z, rw[i].w);
      *(u32x2*)&Ws[buf][(srow + i * 32) * LDX + sc4] = p;
    }
    {
      // transpose: AsT[r][k] = A[k0+srow][r]
      AsT[buf][(sc4 + 0) * LDX + srow] = (uint16_t)bf16_bits(ra.x);
      AsT[buf][(sc4 + 1) * LDX + srow] = (uint16_t)bf16_bits(ra.y);
      AsT[buf][(sc4 + 2) * LDX + srow] = (uint16_t)bf16_bits(ra.z);
      AsT[buf][(sc4 + 3) * LDX + srow] = (uint16_t)bf16_bits(ra.w);
    }

    // ---- issue next K-tile's global loads (overlap with compute below) ----
    if (k0 + BLK_K < DIN) {
      xg += BLK_K;
      wg += BLK_K;
      ag += (size_t)BLK_K * RNK;
      #pragma unroll
      for (int i = 0; i < 8; ++i) rx[i] = *(const f32x4*)(xg + (size_t)(i * 32) * DIN);
      #pragma unroll
      for (int i = 0; i < 4; ++i) rw[i] = *(const f32x4*)(wg + (size_t)(i * 32) * DIN);
      ra = *(const f32x4*)(ag);
      // nudge the tile after next toward L2 while we compute
      __builtin_prefetch(xg + BLK_K, 0, 1);
      __builtin_prefetch(wg + BLK_K, 0, 1);
    }

    __syncthreads();

    // ---- compute: 16 main WMMAs (+8 LoRA WMMAs on nG==0 waves) ----
    v16bf af[4], bfr[4];
    #pragma unroll
    for (int i = 0; i < 4; ++i)
      af[i] = load_a_frag(&Xs[buf][(mG * 64 + i * 16) * LDX], lane);
    #pragma unroll
    for (int j = 0; j < 4; ++j)
      bfr[j] = load_b_frag(&Ws[buf][(nG * 64 + j * 16) * LDX], lane);

    #pragma unroll
    for (int j = 0; j < 4; ++j)
      #pragma unroll
      for (int i = 0; i < 4; ++i)
        acc[i][j] = __builtin_amdgcn_wmma_f32_16x16x32_bf16(
            false, af[i], false, bfr[j], (short)0, acc[i][j], false, false);

    if (nG == 0) {  // wave-uniform branch: accumulate ax = X @ A
      v16bf rf[2];
      rf[0] = load_b_frag(&AsT[buf][0],        lane);
      rf[1] = load_b_frag(&AsT[buf][16 * LDX], lane);
      #pragma unroll
      for (int j = 0; j < 2; ++j)
        #pragma unroll
        for (int i = 0; i < 4; ++i)
          axa[i][j] = __builtin_amdgcn_wmma_f32_16x16x32_bf16(
              false, af[i], false, rf[j], (short)0, axa[i][j], false, false);
    }
  }

  // ---- spill scaled ax to LDS in A-operand (m-major) layout ----
  if (nG == 0) {
    #pragma unroll
    for (int i = 0; i < 4; ++i) {
      #pragma unroll
      for (int j = 0; j < 2; ++j) {
        int col   = j * 16 + (lane & 15);
        int rbase = mG * 64 + i * 16 + ((lane >> 4) << 3);
        #pragma unroll
        for (int v = 0; v < 8; ++v)
          AXs[(rbase + v) * LDX + col] = (uint16_t)bf16_bits(axa[i][j][v] * LSCALE);
      }
    }
  }
  __syncthreads();

  // ---- delta: acc += ax_scaled @ loraB^T (one K=32 WMMA per fragment) ----
  v16bf axf[4];
  #pragma unroll
  for (int i = 0; i < 4; ++i)
    axf[i] = load_a_frag(&AXs[(mG * 64 + i * 16) * LDX], lane);
  #pragma unroll
  for (int j = 0; j < 4; ++j) {
    v16bf blf = load_b_frag(&Bls[(nG * 64 + j * 16) * LDX], lane);
    #pragma unroll
    for (int i = 0; i < 4; ++i)
      acc[i][j] = __builtin_amdgcn_wmma_f32_16x16x32_bf16(
          false, axf[i], false, blf, (short)0, acc[i][j], false, false);
  }

  // ---- epilogue: + bias, store f32 ----
  #pragma unroll
  for (int j = 0; j < 4; ++j) {
    int n = n0 + nG * 64 + j * 16 + (lane & 15);
    float bv = bias[n];
    #pragma unroll
    for (int i = 0; i < 4; ++i) {
      int mb = m0 + mG * 64 + i * 16 + ((lane >> 4) << 3);
      #pragma unroll
      for (int v = 0; v < 8; ++v)
        out[(size_t)(mb + v) * DOUT + n] = acc[i][j][v] + bv;
    }
  }
}

extern "C" void kernel_launch(void* const* d_in, const int* in_sizes, int n_in,
                              void* d_out, int out_size, void* d_ws, size_t ws_size,
                              hipStream_t stream) {
  const float* X    = (const float*)d_in[0];   // [B,S,DIN]
  const float* W    = (const float*)d_in[1];   // [DOUT,DIN]
  const float* bias = (const float*)d_in[2];   // [DOUT]
  const float* lA   = (const float*)d_in[3];   // [NUM_LABELERS,DIN,R]
  const float* lB   = (const float*)d_in[4];   // [DOUT,R]
  const int*   lab  = (const int*)d_in[5];     // [B]
  float* out = (float*)d_out;

  dim3 grid(DOUT / BLK_N, (BATCH * SEQ) / BLK_M);  // (32, 64)
  lora_linear_wmma_kernel<<<grid, 256, 0, stream>>>(X, W, bias, lA, lB, lab, out);
}
